// BertAdapterCapsuleMaskImp_71459665871057
// MI455X (gfx1250) — matrix operand order repaired and checked by reading, not compiled, not run on hardware
//
#include <hip/hip_runtime.h>
#include <math.h>

// MI455X / gfx1250, wave32. WMMA f16 16x16x32 with f32 accumulation for all
// significant GEMMs; fp32 epilogues (exact erf-gelu, sigmoid gates, squash).
// All torch .view reinterpretations are reproduced as exact flat-order
// buffers so the "scrambled" gathers become identity linear reads.
// x is converted fp32->f16 inline while building WMMA B-fragments in k_sem
// (no separate conversion pass; saves ~100 MB of HBM traffic).

typedef _Float16 vh8  __attribute__((ext_vector_type(8)));
typedef _Float16 vh16 __attribute__((ext_vector_type(16)));
typedef float    vf8  __attribute__((ext_vector_type(8)));

#define BB 256
#define SS 128
#define HH 768
#define AA 512
#define NN 10
#define CC 3
#define KK 3
#define MM (BB * SS)   // 32768 token positions

__device__ __forceinline__ float gelu_exact(float x) {
  return 0.5f * x * (1.0f + erff(x * 0.70710678118654752f));
}
__device__ __forceinline__ float sigmoidf_(float x) {
  return 1.0f / (1.0f + expf(-x));
}
__device__ __forceinline__ vf8 wmma_f16(vh16 a, vh16 b, vf8 c) {
  // v_wmma_f32_16x16x32_f16 (8-arg form: neg_a, A, neg_b, B, c_mod, C, reuse_a, reuse_b)
  return __builtin_amdgcn_wmma_f32_16x16x32_f16(false, a, false, b, (short)0, c,
                                                false, false);
}

// A-matrix (16x32 f16, M=row striped in-lane) fragment load.
// Layout: lanes 0-15 hold row M=lane, K = {kb+0..15};
//         lanes 16-31 hold row M=lane-16, K = {kb+16..31}.
// rowk points at &W[row*ldk + kb]; both chunks are contiguous 16B loads.
__device__ __forceinline__ vh16 load_a16x32(const _Float16* __restrict__ rowk,
                                            int lane) {
  const int klo = (lane & 16) ? 16 : 0;
  vh8 c0 = *reinterpret_cast<const vh8*>(rowk + klo);
  vh8 c1 = *reinterpret_cast<const vh8*>(rowk + klo + 8);
  vh16 a;
#pragma unroll
  for (int i = 0; i < 8; i++) { a[i] = c0[i]; a[i + 8] = c1[i]; }
  return a;
}

// ---------------------------------------------------------------- conversions
// Transpose+convert weights to f16 "A-matrix" layout (out-feature major, K contiguous).
__global__ __launch_bounds__(256) void k_cvt_w(
    const float* __restrict__ fc1w, const float* __restrict__ fc2w,
    const float* __restrict__ semw, const float* __restrict__ semb_in,
    _Float16* __restrict__ fc1t, _Float16* __restrict__ fc2t,
    _Float16* __restrict__ semT, float* __restrict__ semb) {
  int i = blockIdx.x * 256 + threadIdx.x;
  if (i < AA * HH) {            // fc1t[n*768 + k] = fc1_w[k*512 + n]
    int n = i / HH, k = i % HH;
    fc1t[i] = (_Float16)fc1w[(size_t)k * AA + n];
  }
  if (i < HH * AA) {            // fc2t[n*512 + k] = fc2_w[k*768 + n]
    int n = i / AA, k = i % AA;
    fc2t[i] = (_Float16)fc2w[(size_t)k * HH + n];
  }
  if (i < 32 * HH) {            // semT[j*768 + k], j = c*10 + n  (cols 30,31 = 0)
    int j = i / HH, k = i % HH;
    _Float16 v = (_Float16)0.0f;
    if (j < 30) {
      int c = j / NN, n = j % NN;
      v = (_Float16)semw[((size_t)n * HH + k) * CC + c];
    }
    semT[i] = v;
  }
  if (i < 32) {
    float v = 0.0f;
    if (i < 30) { int c = i / NN, n = i % NN; v = semb_in[n * CC + c]; }
    semb[i] = v;
  }
}

// ------------------------------------------------ semantic capsules (WMMA + squash)
// One wave per 16-token tile: D tile = semTᵀ(32 x K) x xᵀ(K x 16m).
// B-fragment is built from fp32 x with inline f16 conversion (coalesced b128s).
// u is written in exact (B, S*C, N) flat order: u[m*30 + c*10 + n].
__global__ __launch_bounds__(256) void k_sem(const float* __restrict__ x,
                                             const _Float16* __restrict__ semT,
                                             const float* __restrict__ semb,
                                             float* __restrict__ u) {
  __shared__ float lds[8][16][32];
  const int lane = threadIdx.x & 31;
  const int w    = threadIdx.x >> 5;
  const int m0   = (blockIdx.x * 8 + w) * 16;
  const int arow = lane & 15;
  const int lhi  = lane >> 4;
  const int m    = m0 + arow;

  vf8 acc[2];
#pragma unroll
  for (int i = 0; i < 2; i++)
#pragma unroll
    for (int r = 0; r < 8; r++) acc[i][r] = 0.0f;

  const float*    brow = x + (size_t)m * HH + lhi * 16;
  const _Float16* a0   = semT + (size_t)arow * HH;
  for (int kb = 0; kb < HH; kb += 32) {
    vf8 b0 = *reinterpret_cast<const vf8*>(brow + kb);
    vf8 b1 = *reinterpret_cast<const vf8*>(brow + kb + 8);
    vh16 b;
#pragma unroll
    for (int i = 0; i < 8; i++) {
      b[i]     = (_Float16)b0[i];
      b[i + 8] = (_Float16)b1[i];
    }
    acc[0] = wmma_f16(load_a16x32(a0 + kb, lane), b, acc[0]);
    acc[1] = wmma_f16(load_a16x32(a0 + (size_t)16 * HH + kb, lane), b, acc[1]);
  }
  // stage sem tile (rows=token, cols=j) to LDS for cross-lane squash over tasks
#pragma unroll
  for (int jt = 0; jt < 2; jt++) {
    const int jb = jt * 16 + lhi * 8;
#pragma unroll
    for (int r = 0; r < 8; r++) lds[w][arow][jb + r] = acc[jt][r] + semb[jb + r];
  }
  __syncthreads();
  for (int tt = lane; tt < 48; tt += 32) {   // 16 rows x 3 cap components
    const int c = tt >> 4, row = tt & 15;
    float v[NN];
    float sn = 1e-16f;
#pragma unroll
    for (int n = 0; n < NN; n++) { v[n] = lds[w][row][c * NN + n]; sn += v[n] * v[n]; }
    const float sc = sn / ((1.0f + sn) * sqrtf(sn));
    float* up = u + (size_t)(m0 + row) * 30 + c * NN;
#pragma unroll
    for (int n = 0; n < NN; n++) up[n] = v[n] * sc;
  }
}

// -------------------------------------------------------- dynamic routing (VALU)
// One thread per (k, m). vote stored in exact (K, M, C) flat order.
__global__ __launch_bounds__(256) void k_route(const float* __restrict__ u,
                                               const float* __restrict__ rw,
                                               const int* __restrict__ tptr,
                                               float* __restrict__ vote) {
  const int g = blockIdx.x * 256 + threadIdx.x;
  const int k = g / MM;
  const int m = g % MM;
  const int t = tptr[0];
  const float* um  = u + (size_t)m * 30;        // u[m, n, c] = um[n*3 + c]
  const float* rwk = rw + (size_t)k * (NN * CC * CC);

  float pri[NN][CC];
#pragma unroll
  for (int n = 0; n < NN; n++) {
    const float u0 = um[n * 3 + 0], u1 = um[n * 3 + 1], u2 = um[n * 3 + 2];
    const float* r = rwk + n * 9;
#pragma unroll
    for (int d = 0; d < 3; d++)
      pri[n][d] = u0 * r[d] + u1 * r[3 + d] + u2 * r[6 + d];
  }
  float lg[NN];
#pragma unroll
  for (int n = 0; n < NN; n++) lg[n] = 0.0f;
  float v0 = 0.0f, v1 = 0.0f, v2 = 0.0f;
  for (int it = 0; it < 3; it++) {
#pragma unroll
    for (int n = 0; n < NN; n++) lg[n] = (n <= t) ? lg[n] : -10000.0f;
    float mx = lg[0];
#pragma unroll
    for (int n = 1; n < NN; n++) mx = fmaxf(mx, lg[n]);
    float e[NN], ss = 0.0f;
#pragma unroll
    for (int n = 0; n < NN; n++) { e[n] = expf(lg[n] - mx); ss += e[n]; }
    const float inv = 1.0f / ss;
    v0 = v1 = v2 = 0.0f;
#pragma unroll
    for (int n = 0; n < NN; n++) {
      const float p = e[n] * inv;
      v0 += p * pri[n][0]; v1 += p * pri[n][1]; v2 += p * pri[n][2];
    }
    if (it < 2) {
      const float sn = v0 * v0 + v1 * v1 + v2 * v2 + 1e-16f;
      const float sc = sn / ((1.0f + sn) * sqrtf(sn));
      const float o0 = v0 * sc, o1 = v1 * sc, o2 = v2 * sc;
#pragma unroll
      for (int n = 0; n < NN; n++)
        lg[n] += pri[n][0] * o0 + pri[n][1] * o1 + pri[n][2] * o2;
    }
  }
  float* vp = vote + (size_t)g * 3;   // g = k*MM + m  => (K, M, C) flat
  vp[0] = v0; vp[1] = v1; vp[2] = v2;
}

// --------------------------- capsule head: h@larger_w + gate, h2 = x + ..., -> f16
// h.flat == vote.flat (torch .view), so hv[j] = voteFlat[m*9 + j].
__global__ __launch_bounds__(256) void k_hcap(
    const float* __restrict__ x, const float* __restrict__ vote,
    const float* __restrict__ lw, const float* __restrict__ lb,
    const float* __restrict__ elarger, const int* __restrict__ tptr,
    const int* __restrict__ sptr, _Float16* __restrict__ h2h) {
  __shared__ float hv[9];
  const int m   = blockIdx.y;
  const int col = blockIdx.x * 256 + threadIdx.x;
  if (threadIdx.x < 9) hv[threadIdx.x] = vote[(size_t)m * 9 + threadIdx.x];
  __syncthreads();
  const int t = tptr[0];
  const float sf = (float)sptr[0];
  float acc = lb[col];
#pragma unroll
  for (int j = 0; j < 9; j++) acc += hv[j] * lw[(size_t)j * HH + col];
  const float g  = sigmoidf_(sf * elarger[(size_t)t * HH + col]);
  const float h2 = x[(size_t)m * HH + col] + acc * g;
  h2h[(size_t)m * HH + col] = (_Float16)h2;
}

// ------------------------------------------- fc1: gelu(h2 @ W1 + b1) * gfc1 -> f16
__global__ __launch_bounds__(256) void k_fc1(
    const _Float16* __restrict__ h2h, const _Float16* __restrict__ wT,
    const float* __restrict__ bias, const float* __restrict__ efc,
    const int* __restrict__ tptr, const int* __restrict__ sptr,
    _Float16* __restrict__ outh) {
  const int lane = threadIdx.x & 31;
  const int wid  = blockIdx.x * 8 + (threadIdx.x >> 5);
  const int NTN  = AA / 64;             // 8 n-blocks of 64
  const int m0   = (wid / NTN) * 16;
  const int n0   = (wid % NTN) * 64;
  const int arow = lane & 15;
  const int lhi  = lane >> 4;
  const int m    = m0 + arow;

  vf8 acc[4];
#pragma unroll
  for (int i = 0; i < 4; i++)
#pragma unroll
    for (int r = 0; r < 8; r++) acc[i][r] = 0.0f;

  const _Float16* brow = h2h + (size_t)m * HH + lhi * 16;
  const _Float16* a0   = wT + (size_t)(n0 + arow) * HH;
  for (int kb = 0; kb < HH; kb += 32) {
    vh16 b = *reinterpret_cast<const vh16*>(brow + kb);
#pragma unroll
    for (int nt = 0; nt < 4; nt++)
      acc[nt] = wmma_f16(load_a16x32(a0 + (size_t)(nt * 16) * HH + kb, lane), b,
                         acc[nt]);
  }
  const int t = tptr[0];
  const float sf = (float)sptr[0];
#pragma unroll
  for (int nt = 0; nt < 4; nt++) {
    const int nb = n0 + nt * 16 + lhi * 8;
    vh8 st;
#pragma unroll
    for (int r = 0; r < 8; r++) {
      const int nn = nb + r;
      float v = acc[nt][r] + bias[nn];
      v = gelu_exact(v) * sigmoidf_(sf * efc[(size_t)t * AA + nn]);
      st[r] = (_Float16)v;
    }
    *reinterpret_cast<vh8*>(outh + (size_t)m * AA + nb) = st;
  }
}

// ------------------- fc2: out = x + gelu(a1 @ W2 + b2) * gfc2   (fp32 output)
__global__ __launch_bounds__(256) void k_fc2(
    const _Float16* __restrict__ a1h, const _Float16* __restrict__ wT,
    const float* __restrict__ bias, const float* __restrict__ efc,
    const int* __restrict__ tptr, const int* __restrict__ sptr,
    const float* __restrict__ x, float* __restrict__ out) {
  const int lane = threadIdx.x & 31;
  const int wid  = blockIdx.x * 8 + (threadIdx.x >> 5);
  const int NTN  = HH / 64;             // 12 n-blocks of 64
  const int m0   = (wid / NTN) * 16;
  const int n0   = (wid % NTN) * 64;
  const int arow = lane & 15;
  const int lhi  = lane >> 4;
  const int m    = m0 + arow;

  vf8 acc[4];
#pragma unroll
  for (int i = 0; i < 4; i++)
#pragma unroll
    for (int r = 0; r < 8; r++) acc[i][r] = 0.0f;

  const _Float16* brow = a1h + (size_t)m * AA + lhi * 16;
  const _Float16* a0   = wT + (size_t)(n0 + arow) * AA;
  for (int kb = 0; kb < AA; kb += 32) {
    vh16 b = *reinterpret_cast<const vh16*>(brow + kb);
#pragma unroll
    for (int nt = 0; nt < 4; nt++)
      acc[nt] = wmma_f16(load_a16x32(a0 + (size_t)(nt * 16) * AA + kb, lane), b,
                         acc[nt]);
  }
  const int t = tptr[0];
  const float sf = (float)sptr[0];
#pragma unroll
  for (int nt = 0; nt < 4; nt++) {
    const int nb = n0 + nt * 16 + lhi * 8;
    vf8 xv = *reinterpret_cast<const vf8*>(x + (size_t)m * HH + nb);
    vf8 st;
#pragma unroll
    for (int r = 0; r < 8; r++) {
      const int nn = nb + r;
      float v = acc[nt][r] + bias[nn];
      v = gelu_exact(v) * sigmoidf_(sf * efc[(size_t)t * HH + nn]);
      st[r] = xv[r] + v;
    }
    *reinterpret_cast<vf8*>(out + (size_t)m * HH + nb) = st;
  }
}

extern "C" void kernel_launch(void* const* d_in, const int* in_sizes, int n_in,
                              void* d_out, int out_size, void* d_ws,
                              size_t ws_size, hipStream_t stream) {
  (void)in_sizes; (void)n_in; (void)out_size; (void)ws_size;
  const float* x       = (const float*)d_in[0];
  const int*   tptr    = (const int*)d_in[1];
  const int*   sptr    = (const int*)d_in[2];
  const float* fc1w    = (const float*)d_in[3];
  const float* fc1b    = (const float*)d_in[4];
  const float* fc2w    = (const float*)d_in[5];
  const float* fc2b    = (const float*)d_in[6];
  const float* efc1    = (const float*)d_in[7];
  const float* efc2    = (const float*)d_in[8];
  const float* semw    = (const float*)d_in[9];
  const float* semb_i  = (const float*)d_in[10];
  const float* routew  = (const float*)d_in[11];
  const float* largerw = (const float*)d_in[12];
  const float* largerb = (const float*)d_in[13];
  const float* elarger = (const float*)d_in[14];
  float* out = (float*)d_out;

  // Workspace partition (~91 MB, 256B aligned slabs)
  char* p = (char*)d_ws;
  auto take = [&](size_t bytes) -> char* {
    char* r = p;
    p += (bytes + 255) & ~size_t(255);
    return r;
  };
  _Float16* h2h  = (_Float16*)take((size_t)MM * HH * 2);  // x + capsule head, f16
  _Float16* a1h  = (_Float16*)take((size_t)MM * AA * 2);  // fc1 activations, f16
  _Float16* fc1t = (_Float16*)take((size_t)AA * HH * 2);  // W1^T f16
  _Float16* fc2t = (_Float16*)take((size_t)HH * AA * 2);  // W2^T f16
  _Float16* semT = (_Float16*)take((size_t)32 * HH * 2);  // sem W^T f16 (padded to 32)
  float*    semb = (float*)take(32 * 4);
  float*    u    = (float*)take((size_t)MM * 30 * 4);     // squashed sem caps
  float*    vote = (float*)take((size_t)KK * MM * CC * 4);

  k_cvt_w<<<(HH * AA + 255) / 256, 256, 0, stream>>>(
      fc1w, fc2w, semw, semb_i, fc1t, fc2t, semT, semb);
  k_sem<<<MM / 16 / 8, 256, 0, stream>>>(x, semT, semb, u);
  k_route<<<KK * MM / 256, 256, 0, stream>>>(u, routew, tptr, vote);
  k_hcap<<<dim3(HH / 256, MM), 256, 0, stream>>>(x, vote, largerw, largerb,
                                                 elarger, tptr, sptr, h2h);
  k_fc1<<<(AA / 64) * (MM / 16) / 8, 256, 0, stream>>>(h2h, fc1t, fc1b, efc1,
                                                       tptr, sptr, a1h);
  k_fc2<<<(HH / 64) * (MM / 16) / 8, 256, 0, stream>>>(a1h, fc2t, fc2b, efc2,
                                                       tptr, sptr, x, out);
}
